// CrossModalTransformerLayer_22995254903113
// MI455X (gfx1250) — matrix operand order repaired and checked by reading, not compiled, not run on hardware
//
#include <hip/hip_runtime.h>
#include <cstdint>
#include <cstddef>

#define E_DIM 1024
#define B_ROWS 8192

typedef __attribute__((ext_vector_type(16))) __bf16 v16bf;
typedef __attribute__((ext_vector_type(8)))  float  v8f;
typedef __attribute__((ext_vector_type(4)))  unsigned int v4u;
typedef unsigned short ushort_t;

// ---------------------------------------------------------------- helpers ---

__device__ __forceinline__ unsigned short f2bf(float x) {
  unsigned int u = __float_as_uint(x);
  u += 0x7fffu + ((u >> 16) & 1u);   // round-to-nearest-even to bf16
  return (unsigned short)(u >> 16);
}

__device__ __forceinline__ float gelu_exact(float x) {
  return 0.5f * x * (1.0f + erff(x * 0.70710678118654752440f));
}

// Fragment set for one K=32 step of the 32xN_wave tile.
struct Frags {
  v16bf a0, a1;   // A rows mbase..mbase+15 / mbase+16..mbase+31 (per ISA A layout)
  v16bf b[8];     // 8 N-tiles of B (W rows, K-contiguous)
};

// A fragment: 16x32 bf16, row M = lane%16.
// lanes 0-15 hold K=kk+[0..7],kk+[16..23]; lanes 16-31 hold kk+[8..15],kk+[24..31].
__device__ __forceinline__ v16bf load_a(const ushort_t* p /*row base + lane koff*/, int kk) {
  union { v16bf v; v4u u[2]; } f;
  f.u[0] = *reinterpret_cast<const v4u*>(p + kk);
  f.u[1] = *reinterpret_cast<const v4u*>(p + kk + 16);
  return f.v;
}

// B fragment: 32x16 bf16 (KxN), col N = lane%16 == row of W[N,K].
// lanes 0-15: K=kk+[0..7],kk+[16..23]; lanes 16-31: K=kk+[8..15],kk+[24..31].
__device__ __forceinline__ v16bf load_b(const ushort_t* p /*row base + lane koff*/, int kk) {
  union { v16bf v; v4u u[2]; } f;
  f.u[0] = *reinterpret_cast<const v4u*>(p + kk);
  f.u[1] = *reinterpret_cast<const v4u*>(p + kk + 16);
  return f.v;
}

// ---------------------------------------------------------------- kernels ---

__global__ __launch_bounds__(256) void cvt_f32_bf16(const float* __restrict__ in,
                                                    ushort_t* __restrict__ out, int n) {
  int i = blockIdx.x * blockDim.x + threadIdx.x;
  if (i < n) out[i] = f2bf(in[i]);
}

// MODE 0: out_bf16 = acc + bias
// MODE 1: out_bf16 = gelu(acc + bias)
// MODE 2: out_f32 (+opt out_bf16) = LayerNorm(resid + acc + bias); N==1024, gridDim.y==1
template <int MODE>
__global__ __launch_bounds__(256) void gemm_bf16_ep(
    const ushort_t* __restrict__ X,    // [M,K] bf16
    const ushort_t* __restrict__ Wt,   // [N,K] bf16 (reference does x @ W.T)
    const float*    __restrict__ bias, // [N]
    const float*    __restrict__ resid,// [M,N] f32 (MODE 2)
    const float*    __restrict__ lnw,  // [N]   (MODE 2)
    const float*    __restrict__ lnb,  // [N]   (MODE 2)
    float*          __restrict__ outf, // [M,N] f32 (MODE 2)
    ushort_t*       __restrict__ outb, // [M,N] bf16 (MODE 0/1, optional MODE 2)
    int K, int N) {
  __shared__ float s_sum[32];
  __shared__ float s_sq[32];

  const int lane  = threadIdx.x & 31;
  const int wave  = threadIdx.x >> 5;
  const int mbase = blockIdx.x * 32;
  const int nwave = blockIdx.y * 1024 + wave * 128;
  const int ll = lane & 15;
  const int lh = lane >> 4;

  // Hoisted per-lane row base pointers (include the lane's K sub-offset).
  const int koff_a = lh << 3;   // 0 or 8
  const int koff_b = lh << 3;   // 0 or 8
  const ushort_t* pa0 = X + (size_t)(mbase + ll) * (size_t)K + koff_a;
  const ushort_t* pa1 = pa0 + (size_t)16 * (size_t)K;
  const ushort_t* pb[8];
#pragma unroll
  for (int t = 0; t < 8; ++t)
    pb[t] = Wt + (size_t)(nwave + 16 * t + ll) * (size_t)K + koff_b;

  v8f acc[2][8];
#pragma unroll
  for (int mt = 0; mt < 2; ++mt)
#pragma unroll
    for (int t = 0; t < 8; ++t)
      acc[mt][t] = (v8f){0.f, 0.f, 0.f, 0.f, 0.f, 0.f, 0.f, 0.f};

  auto load_frags = [&](Frags& f, int kk) {
    f.a0 = load_a(pa0, kk);
    f.a1 = load_a(pa1, kk);
#pragma unroll
    for (int t = 0; t < 8; ++t) f.b[t] = load_b(pb[t], kk);
  };
  auto mma_frags = [&](const Frags& f) {
#pragma unroll
    for (int t = 0; t < 8; ++t) {
      acc[0][t] = __builtin_amdgcn_wmma_f32_16x16x32_bf16(
          false, f.a0, false, f.b[t], (short)0, acc[0][t], false, false);
      acc[1][t] = __builtin_amdgcn_wmma_f32_16x16x32_bf16(
          false, f.a1, false, f.b[t], (short)0, acc[1][t], false, false);
    }
  };

  // Double-buffered pipeline; K is a multiple of 64 (1024 or 4096).
  Frags f0, f1;
  load_frags(f0, 0);
  for (int kk = 0; kk < K; kk += 64) {
    load_frags(f1, kk + 32);      // prefetch step k+32 while consuming k
    mma_frags(f0);
    if (kk + 64 < K) {
      load_frags(f0, kk + 64);    // prefetch step k+64 while consuming k+32
    }
    mma_frags(f1);
  }

  // per-column epilogue constants
  float bi[8], lw[8], lb[8];
#pragma unroll
  for (int t = 0; t < 8; ++t) {
    int n = nwave + 16 * t + ll;
    bi[t] = bias[n];
    if (MODE == 2) { lw[t] = lnw[n]; lb[t] = lnb[n]; }
  }

  if (MODE == 0 || MODE == 1) {
#pragma unroll
    for (int mt = 0; mt < 2; ++mt)
#pragma unroll
      for (int t = 0; t < 8; ++t)
#pragma unroll
        for (int r = 0; r < 8; ++r) {
          int m = mbase + 16 * mt + 8 * lh + r;
          int n = nwave + 16 * t + ll;
          float v = acc[mt][t][r] + bi[t];
          if (MODE == 1) v = gelu_exact(v);
          outb[(size_t)m * (size_t)N + (size_t)n] = f2bf(v);
        }
    return;
  }

  // MODE 2: residual add, then LayerNorm over the full row (N == 1024)
  if (threadIdx.x < 32) { s_sum[threadIdx.x] = 0.f; s_sq[threadIdx.x] = 0.f; }
  __syncthreads();

#pragma unroll
  for (int mt = 0; mt < 2; ++mt)
#pragma unroll
    for (int t = 0; t < 8; ++t)
#pragma unroll
      for (int r = 0; r < 8; ++r) {
        int m = mbase + 16 * mt + 8 * lh + r;
        int n = nwave + 16 * t + ll;
        acc[mt][t][r] += bi[t] + resid[(size_t)m * (size_t)N + (size_t)n];
      }

  // row partial sums over this wave's 128 columns
#pragma unroll
  for (int mt = 0; mt < 2; ++mt)
#pragma unroll
    for (int r = 0; r < 8; ++r) {
      float ps = 0.f, sq = 0.f;
#pragma unroll
      for (int t = 0; t < 8; ++t) {
        float v = acc[mt][t][r];
        ps += v; sq += v * v;
      }
#pragma unroll
      for (int off = 8; off > 0; off >>= 1) {
        ps += __shfl_xor(ps, off, 16);
        sq += __shfl_xor(sq, off, 16);
      }
      if (ll == 0) {
        int mloc = 16 * mt + 8 * lh + r;
        atomicAdd(&s_sum[mloc], ps);   // ds_add_f32
        atomicAdd(&s_sq[mloc],  sq);
      }
    }
  __syncthreads();

  const float invN = 1.0f / 1024.0f;
#pragma unroll
  for (int mt = 0; mt < 2; ++mt)
#pragma unroll
    for (int r = 0; r < 8; ++r) {
      int mloc = 16 * mt + 8 * lh + r;
      float mean = s_sum[mloc] * invN;
      float var  = s_sq[mloc] * invN - mean * mean;
      float rstd = rsqrtf(var + 1e-5f);
      int m = mbase + mloc;
#pragma unroll
      for (int t = 0; t < 8; ++t) {
        int n = nwave + 16 * t + ll;
        float o = (acc[mt][t][r] - mean) * rstd * lw[t] + lb[t];
        outf[(size_t)m * (size_t)N + (size_t)n] = o;
        if (outb) outb[(size_t)m * (size_t)N + (size_t)n] = f2bf(o);
      }
    }
}

// ----------------------------------------------------------------- launch ---

extern "C" void kernel_launch(void* const* d_in, const int* in_sizes, int n_in,
                              void* d_out, int out_size, void* d_ws, size_t ws_size,
                              hipStream_t stream) {
  (void)in_sizes; (void)n_in; (void)out_size; (void)ws_size;

  const float* drug = (const float*)d_in[0];
  const float* se   = (const float*)d_in[1];
  // drug-side params (Wq/Wk/bq/bk dead: softmax over 1 key == 1.0)
  const float* Wv_d = (const float*)d_in[4];
  const float* Wo_d = (const float*)d_in[5];
  const float* bv_d = (const float*)d_in[8];
  const float* bo_d = (const float*)d_in[9];
  const float* n1w_d = (const float*)d_in[10];
  const float* n1b_d = (const float*)d_in[11];
  const float* W1_d = (const float*)d_in[12];
  const float* b1_d = (const float*)d_in[13];
  const float* W2_d = (const float*)d_in[14];
  const float* b2_d = (const float*)d_in[15];
  const float* lnw_d = (const float*)d_in[16];
  const float* lnb_d = (const float*)d_in[17];
  // se-side params
  const float* Wv_s = (const float*)d_in[20];
  const float* Wo_s = (const float*)d_in[21];
  const float* bv_s = (const float*)d_in[24];
  const float* bo_s = (const float*)d_in[25];
  const float* n1w_s = (const float*)d_in[26];
  const float* n1b_s = (const float*)d_in[27];
  const float* W1_s = (const float*)d_in[28];
  const float* b1_s = (const float*)d_in[29];
  const float* W2_s = (const float*)d_in[30];
  const float* b2_s = (const float*)d_in[31];
  const float* lnw_s = (const float*)d_in[32];
  const float* lnb_s = (const float*)d_in[33];

  const size_t BE = (size_t)B_ROWS * E_DIM;         // 8.4M
  const size_t EE = (size_t)E_DIM * E_DIM;          // 1M
  const size_t BH = (size_t)B_ROWS * 4 * E_DIM;     // 33.5M

  char* wsp = (char*)d_ws;
  auto carve = [&](size_t bytes) -> void* {
    void* p = (void*)wsp;
    wsp += (bytes + 255) & ~(size_t)255;
    return p;
  };
  ushort_t* xd   = (ushort_t*)carve(BE * 2);
  ushort_t* xs   = (ushort_t*)carve(BE * 2);
  ushort_t* wvd  = (ushort_t*)carve(EE * 2);
  ushort_t* wod  = (ushort_t*)carve(EE * 2);
  ushort_t* w1d  = (ushort_t*)carve(EE * 8);
  ushort_t* w2d  = (ushort_t*)carve(EE * 8);
  ushort_t* wvs  = (ushort_t*)carve(EE * 2);
  ushort_t* wos  = (ushort_t*)carve(EE * 2);
  ushort_t* w1s  = (ushort_t*)carve(EE * 8);
  ushort_t* w2s  = (ushort_t*)carve(EE * 8);
  ushort_t* tmpv = (ushort_t*)carve(BE * 2);
  float*    x1f  = (float*)carve(BE * 4);
  ushort_t* x1b  = (ushort_t*)carve(BE * 2);
  ushort_t* hb   = (ushort_t*)carve(BH * 2);
  ushort_t* dbf  = (ushort_t*)carve(BE * 2);

  dim3 blk(256);
  auto cvt = [&](const float* src, ushort_t* dst, size_t n) {
    cvt_f32_bf16<<<dim3((unsigned)((n + 255) / 256)), blk, 0, stream>>>(src, dst, (int)n);
  };
  cvt(drug, xd, BE);
  cvt(se,   xs, BE);
  cvt(Wv_d, wvd, EE); cvt(Wo_d, wod, EE); cvt(W1_d, w1d, 4 * EE); cvt(W2_d, w2d, 4 * EE);
  cvt(Wv_s, wvs, EE); cvt(Wo_s, wos, EE); cvt(W1_s, w1s, 4 * EE); cvt(W2_s, w2s, 4 * EE);

  dim3 gE(B_ROWS / 32, 1);   // N = 1024
  dim3 gH(B_ROWS / 32, 4);   // N = 4096

  float* out_d = (float*)d_out;
  float* out_s = out_d + BE;

  // ---- drug attends to SE ----
  gemm_bf16_ep<0><<<gE, blk, 0, stream>>>(xs, wvd, bv_d, nullptr, nullptr, nullptr,
                                          nullptr, tmpv, E_DIM, E_DIM);
  gemm_bf16_ep<2><<<gE, blk, 0, stream>>>(tmpv, wod, bo_d, drug, n1w_d, n1b_d,
                                          x1f, x1b, E_DIM, E_DIM);
  gemm_bf16_ep<1><<<gH, blk, 0, stream>>>(x1b, w1d, b1_d, nullptr, nullptr, nullptr,
                                          nullptr, hb, E_DIM, 4 * E_DIM);
  gemm_bf16_ep<2><<<gE, blk, 0, stream>>>(hb, w2d, b2_d, x1f, lnw_d, lnb_d,
                                          out_d, dbf, 4 * E_DIM, E_DIM);
  // ---- SE attends to updated drug ----
  gemm_bf16_ep<0><<<gE, blk, 0, stream>>>(dbf, wvs, bv_s, nullptr, nullptr, nullptr,
                                          nullptr, tmpv, E_DIM, E_DIM);
  gemm_bf16_ep<2><<<gE, blk, 0, stream>>>(tmpv, wos, bo_s, se, n1w_s, n1b_s,
                                          x1f, x1b, E_DIM, E_DIM);
  gemm_bf16_ep<1><<<gH, blk, 0, stream>>>(x1b, w1s, b1_s, nullptr, nullptr, nullptr,
                                          nullptr, hb, E_DIM, 4 * E_DIM);
  gemm_bf16_ep<2><<<gE, blk, 0, stream>>>(hb, w2s, b2_s, x1f, lnw_s, lnb_s,
                                          out_s, nullptr, 4 * E_DIM, E_DIM);
}
